// HGNN_82197084111196
// MI455X (gfx1250) — compile-verified, matrix operand order checked by reading, not run on hardware
//
#include <hip/hip_runtime.h>
#include <math.h>

// ---------------- constants ----------------
#define Nn    10000
#define NPAD  10016          // multiple of 16
#define Ee    150000
#define F0c   128
#define Hh    8
#define Dd    64
#define HDc   512
#define Gg    64
#define Cc    10
#define SLOPE 0.01f
#define EPSb  1e-5f

typedef __attribute__((ext_vector_type(16))) __bf16 v16bf;
typedef __attribute__((ext_vector_type(8)))  float  v8f;

// ---------------- helpers ----------------
__device__ __forceinline__ unsigned short f2bf(float f) {
    unsigned int u = __float_as_uint(f);
    u = (u + 0x7FFFu + ((u >> 16) & 1u)) >> 16;     // round-to-nearest-even
    return (unsigned short)u;
}

__device__ __forceinline__ void atomicMaxF(float* addr, float val) {
    unsigned int* ua = (unsigned int*)addr;
    unsigned int old = *ua;
    while (true) {
        float f = __uint_as_float(old);
        if (f >= val) break;
        unsigned int assumed = old;
        old = atomicCAS(ua, assumed, __float_as_uint(val));
        if (old == assumed) break;
    }
}

// ---------------- conversion kernels ----------------
// x [nrows,K] f32 -> xb [npad,K] bf16, zero-padded rows
__global__ void k_cvt_a(const float* __restrict__ x, unsigned short* __restrict__ xb,
                        int nrows, int K, long total) {
    long i = (long)blockIdx.x * blockDim.x + threadIdx.x;
    if (i >= total) return;
    long r = i / K;
    float v = (r < nrows) ? x[i] : 0.f;
    xb[i] = f2bf(v);
}

// w [K,Nc] f32 -> wt [Nc,K] bf16 (transposed)
__global__ void k_cvt_bt(const float* __restrict__ w, unsigned short* __restrict__ wt,
                         int K, int Nc) {
    long i = (long)blockIdx.x * blockDim.x + threadIdx.x;
    if (i >= (long)K * Nc) return;
    int k = (int)(i / Nc), n = (int)(i % Nc);
    wt[(long)n * K + k] = f2bf(w[i]);
}

// ---------------- WMMA GEMM: C[M,Nc] = A[M,K] * B[K,Nc], B given transposed ----------------
__global__ void __launch_bounds__(256)
k_gemm_bf16_wmma(const unsigned short* __restrict__ A, const unsigned short* __restrict__ Bt,
                 float* __restrict__ C, int M, int K, int Nc, int T) {
    int wave = blockIdx.x * (blockDim.x >> 5) + (threadIdx.x >> 5);
    int lane = threadIdx.x & 31;
    if (wave >= T) return;                       // wave-uniform: EXEC stays all-1 below
    int ntn  = Nc >> 4;
    int tm   = wave / ntn, tn = wave % ntn;
    int r    = lane & 15;
    int kg   = (lane >> 4) * 8;                  // 0 or 8
    const unsigned short* arow = A  + (long)(tm * 16 + r) * K;
    const unsigned short* brow = Bt + (long)(tn * 16 + r) * K;
    v8f acc = {};
    for (int k0 = 0; k0 < K; k0 += 32) {
        union { v16bf v; uint4 q[2]; } a, b;
        a.q[0] = *(const uint4*)(arow + k0 + kg);
        a.q[1] = *(const uint4*)(arow + k0 + 16 + kg);
        b.q[0] = *(const uint4*)(brow + k0 + kg);
        b.q[1] = *(const uint4*)(brow + k0 + 16 + kg);
        acc = __builtin_amdgcn_wmma_f32_16x16x32_bf16(
                  false, a.v, false, b.v, (short)0, acc, false, false);
    }
    // D layout: lanes 0-15 -> rows 0..7 (col=lane); lanes 16-31 -> rows 8..15 (col=lane-16)
    int col   = lane & 15;
    int rbase = (lane >> 4) * 8;
    float* cp = C + (long)(tm * 16 + rbase) * Nc + tn * 16 + col;
#pragma unroll
    for (int v = 0; v < 8; ++v) cp[(long)v * Nc] = acc[v];
}

// ---------------- attention scores ----------------
__global__ void k_scores(const float* __restrict__ h, const float* __restrict__ as_,
                         const float* __restrict__ ad_, float* __restrict__ es,
                         float* __restrict__ ed, int total) {
    int i = blockIdx.x * blockDim.x + threadIdx.x;
    if (i >= total) return;
    int node = i / Hh, hh = i % Hh;
    const float* hp = h + (long)node * HDc + hh * Dd;
    const float* ap = as_ + hh * Dd;
    const float* dp = ad_ + hh * Dd;
    float s = 0.f, d = 0.f;
#pragma unroll 8
    for (int k = 0; k < Dd; ++k) { float v = hp[k]; s += v * ap[k]; d += v * dp[k]; }
    es[i] = s; ed[i] = d;
}

// ---------------- per-layer init ----------------
__global__ void k_init_layer(float* __restrict__ agg, float* __restrict__ m,
                             float* __restrict__ denom, long nagg, int nh) {
    long i = (long)blockIdx.x * blockDim.x + threadIdx.x;
    if (i < nagg) agg[i] = 0.f;
    if (i < nh) { m[i] = -INFINITY; denom[i] = 0.f; }
}

// ---------------- edge phase ----------------
__global__ void k_edge_max(const int* __restrict__ src, const int* __restrict__ dst,
                           const float* __restrict__ es, const float* __restrict__ ed,
                           float* __restrict__ elr, float* __restrict__ m, int total) {
    int i = blockIdx.x * blockDim.x + threadIdx.x;
    if (i >= total) return;
    int e = i / Hh, hh = i % Hh;
    int s = src[e], d = dst[e];
    float v = es[s * Hh + hh] + ed[d * Hh + hh];
    v = v > 0.f ? v : SLOPE * v;
    elr[i] = v;
    atomicMaxF(&m[d * Hh + hh], v);
}

__global__ void k_edge_exp(const int* __restrict__ dst, const float* __restrict__ elr,
                           const float* __restrict__ m, float* __restrict__ exb,
                           float* __restrict__ denom, int total) {
    int i = blockIdx.x * blockDim.x + threadIdx.x;
    if (i >= total) return;
    int e = i / Hh, hh = i % Hh;
    int d = dst[e];
    float x = expf(elr[i] - m[d * Hh + hh]);
    exb[i] = x;
    atomicAdd(&denom[d * Hh + hh], x);
}

// one block (128 thr) per edge; each thread covers 4 features
__global__ void k_edge_agg(const int* __restrict__ src, const int* __restrict__ dst,
                           const float* __restrict__ h, const float* __restrict__ exb,
                           const float* __restrict__ denom, float* __restrict__ agg) {
    int e = blockIdx.x;
    int t = threadIdx.x;
    int s = src[e], d = dst[e];
#pragma unroll
    for (int i = 0; i < 4; ++i) {
        int f  = t + i * 128;
        int hh = f >> 6;
        float att = exb[e * Hh + hh] / denom[d * Hh + hh];
        float v   = h[(long)s * HDc + f] * att;
        atomicAdd(&agg[(long)d * HDc + f], v);
    }
}

__global__ void k_bias_elu(const float* __restrict__ agg, const float* __restrict__ b,
                           float* __restrict__ xout, long total) {
    long i = (long)blockIdx.x * blockDim.x + threadIdx.x;
    if (i >= total) return;
    int f = (int)(i % HDc);
    float v = agg[i] + b[f];
    xout[i] = v > 0.f ? v : (expf(v) - 1.f);
}

// ---------------- misc elementwise ----------------
__global__ void k_residual(const float* __restrict__ a, const float* __restrict__ b,
                           float* __restrict__ c, long total) {
    long i = (long)blockIdx.x * blockDim.x + threadIdx.x;
    if (i < total) c[i] = a[i] + b[i];
}

__global__ void k_fill(float* __restrict__ p, float val, long total) {
    long i = (long)blockIdx.x * blockDim.x + threadIdx.x;
    if (i < total) p[i] = val;
}

// ---------------- pooling ----------------
__global__ void k_pool_max(const int* __restrict__ batch,
                           const float* __restrict__ xt, const float* __restrict__ x1,
                           const float* __restrict__ x2, const float* __restrict__ x3,
                           float* __restrict__ pooled, long total) {
    long i = (long)blockIdx.x * blockDim.x + threadIdx.x;
    if (i >= total) return;
    int node = (int)(i / (4 * HDc));
    int f    = (int)(i % (4 * HDc));
    float v;
    if      (f < HDc)     v = xt[(long)node * HDc + f];
    else if (f < 2 * HDc) v = x1[(long)node * HDc + f - HDc];
    else if (f < 3 * HDc) v = x2[(long)node * HDc + f - 2 * HDc];
    else                  v = x3[(long)node * HDc + f - 3 * HDc];
    int g = batch[node];
    atomicMaxF(&pooled[(long)g * (4 * HDc) + f], v);
}

__global__ void k_pool_fin(float* __restrict__ pooled, int total) {
    int i = blockIdx.x * blockDim.x + threadIdx.x;
    if (i >= total) return;
    float v = pooled[i];
    if (!isfinite(v)) pooled[i] = 0.f;
}

// ---------------- MLP tail ----------------
__global__ void k_bn_relu(float* __restrict__ z, const float* __restrict__ mb1,
                          const float* __restrict__ gamma, const float* __restrict__ beta) {
    int j = blockIdx.x * blockDim.x + threadIdx.x;
    if (j >= HDc) return;
    float b = mb1[j];
    float mu = 0.f;
    for (int g = 0; g < Gg; ++g) mu += z[g * HDc + j] + b;
    mu *= (1.f / Gg);
    float var = 0.f;
    for (int g = 0; g < Gg; ++g) { float d = z[g * HDc + j] + b - mu; var += d * d; }
    var *= (1.f / Gg);
    float inv = rsqrtf(var + EPSb);
    for (int g = 0; g < Gg; ++g) {
        float v = (z[g * HDc + j] + b - mu) * inv * gamma[j] + beta[j];
        z[g * HDc + j] = fmaxf(v, 0.f);
    }
}

__global__ void k_mlp2(const float* __restrict__ z, const float* __restrict__ mW2,
                       const float* __restrict__ mb2, float* __restrict__ out, int total) {
    int i = blockIdx.x * blockDim.x + threadIdx.x;
    if (i >= total) return;
    int g = i / Cc, c = i % Cc;
    float s = mb2[c];
    for (int k = 0; k < HDc; ++k) s += z[g * HDc + k] * mW2[k * Cc + c];
    out[i] = s;
}

// ---------------- host-side orchestration ----------------
static inline int cdiv(long a, long b) { return (int)((a + b - 1) / b); }

struct LayerWs {
    unsigned short* xb;   // bf16 A staging
    unsigned short* wt;   // bf16 transposed B staging
    float* buf_h;         // [NPAD, HD] pre-aggregation transform
    float* agg;           // [N, HD]
    float *es, *ed, *m, *denom;
    float* exb;           // [E, H]
};

static void run_gat(const float* x_in, int Fin,
                    const float* W, const float* as_, const float* ad_, const float* b,
                    const int* src, const int* dst,
                    float* x_out, const LayerWs& w, hipStream_t stream) {
    long aTot = (long)NPAD * Fin;
    k_cvt_a<<<cdiv(aTot, 256), 256, 0, stream>>>(x_in, w.xb, Nn, Fin, aTot);
    k_cvt_bt<<<cdiv((long)Fin * HDc, 256), 256, 0, stream>>>(W, w.wt, Fin, HDc);
    int T = (NPAD / 16) * (HDc / 16);
    k_gemm_bf16_wmma<<<cdiv(T, 8), 256, 0, stream>>>(w.xb, w.wt, w.buf_h, NPAD, Fin, HDc, T);
    k_scores<<<cdiv(Nn * Hh, 256), 256, 0, stream>>>(w.buf_h, as_, ad_, w.es, w.ed, Nn * Hh);
    long nagg = (long)Nn * HDc;
    k_init_layer<<<cdiv(nagg, 256), 256, 0, stream>>>(w.agg, w.m, w.denom, nagg, Nn * Hh);
    k_edge_max<<<cdiv(Ee * Hh, 256), 256, 0, stream>>>(src, dst, w.es, w.ed, w.exb, w.m, Ee * Hh);
    k_edge_exp<<<cdiv(Ee * Hh, 256), 256, 0, stream>>>(dst, w.exb, w.m, w.exb, w.denom, Ee * Hh);
    k_edge_agg<<<Ee, 128, 0, stream>>>(src, dst, w.buf_h, w.exb, w.denom, w.agg);
    k_bias_elu<<<cdiv(nagg, 256), 256, 0, stream>>>(w.agg, b, x_out, nagg);
}

extern "C" void kernel_launch(void* const* d_in, const int* in_sizes, int n_in,
                              void* d_out, int out_size, void* d_ws, size_t ws_size,
                              hipStream_t stream) {
    // inputs (setup_inputs order)
    const float* x    = (const float*)d_in[0];
    const int*   eidx = (const int*)d_in[1];
    const int*   batch= (const int*)d_in[2];
    const float *W0=(const float*)d_in[3],  *as0=(const float*)d_in[4],
                *ad0=(const float*)d_in[5], *b0 =(const float*)d_in[6];
    const float *W1=(const float*)d_in[7],  *as1=(const float*)d_in[8],
                *ad1=(const float*)d_in[9], *b1 =(const float*)d_in[10];
    const float *W2=(const float*)d_in[11], *as2=(const float*)d_in[12],
                *ad2=(const float*)d_in[13],*b2 =(const float*)d_in[14];
    const float *W3=(const float*)d_in[15], *as3=(const float*)d_in[16],
                *ad3=(const float*)d_in[17],*b3 =(const float*)d_in[18];
    const float *mW1=(const float*)d_in[19],*mb1=(const float*)d_in[20];
    const float *gamma=(const float*)d_in[21],*beta=(const float*)d_in[22];
    const float *mW2=(const float*)d_in[23],*mb2=(const float*)d_in[24];
    float* out = (float*)d_out;

    const int* src = eidx;
    const int* dst = eidx + Ee;

    // workspace carve-up (256B aligned)
    size_t off = 0;
    char* base = (char*)d_ws;
    auto alloc = [&](size_t bytes) -> char* {
        char* p = base + off;
        off += (bytes + 255) & ~(size_t)255;
        return p;
    };
    LayerWs w;
    w.xb    = (unsigned short*)alloc((size_t)NPAD * HDc * 2);     // covers 64x2048 MLP A too
    w.wt    = (unsigned short*)alloc((size_t)HDc * (4 * HDc) * 2);// covers [512,2048] transposed mW1
    w.buf_h = (float*)alloc((size_t)NPAD * HDc * 4);
    w.agg   = (float*)alloc((size_t)Nn * HDc * 4);
    float* x0 = (float*)alloc((size_t)Nn * HDc * 4);
    float* x1 = (float*)alloc((size_t)Nn * HDc * 4);
    float* x2 = (float*)alloc((size_t)Nn * HDc * 4);
    float* x3 = (float*)alloc((size_t)Nn * HDc * 4);
    float* xt = (float*)alloc((size_t)Nn * HDc * 4);
    w.es    = (float*)alloc((size_t)Nn * Hh * 4);
    w.ed    = (float*)alloc((size_t)Nn * Hh * 4);
    w.m     = (float*)alloc((size_t)Nn * Hh * 4);
    w.denom = (float*)alloc((size_t)Nn * Hh * 4);
    w.exb   = (float*)alloc((size_t)Ee * Hh * 4);
    float* pooled = (float*)alloc((size_t)Gg * 4 * HDc * 4);
    float* z      = (float*)alloc((size_t)Gg * HDc * 4);

    // ---- four GAT layers ----
    run_gat(x,  F0c, W0, as0, ad0, b0, src, dst, x0, w, stream);   // init_graph
    run_gat(x0, HDc, W1, as1, ad1, b1, src, dst, x1, w, stream);   // head
    run_gat(x1, HDc, W2, as2, ad2, b2, src, dst, x2, w, stream);   // body
    long nt = (long)Nn * HDc;
    k_residual<<<cdiv(nt, 256), 256, 0, stream>>>(x0, x2, x3, nt); // x3 = x0 + x2
    run_gat(x3, HDc, W3, as3, ad3, b3, src, dst, xt, w, stream);   // tail

    // ---- pooling: segment_max over batch of concat(xt,x1,x2,x3) ----
    long pTot = (long)Gg * 4 * HDc;
    k_fill<<<cdiv(pTot, 256), 256, 0, stream>>>(pooled, -INFINITY, pTot);
    long pmTot = (long)Nn * 4 * HDc;
    k_pool_max<<<cdiv(pmTot, 256), 256, 0, stream>>>(batch, xt, x1, x2, x3, pooled, pmTot);
    k_pool_fin<<<cdiv(pTot, 256), 256, 0, stream>>>(pooled, (int)pTot);

    // ---- MLP: Linear (via WMMA) -> BN -> ReLU -> Linear ----
    long aTot = (long)Gg * 4 * HDc;   // 64 rows (multiple of 16), K=2048
    k_cvt_a<<<cdiv(aTot, 256), 256, 0, stream>>>(pooled, w.xb, Gg, 4 * HDc, aTot);
    k_cvt_bt<<<cdiv((long)4 * HDc * HDc, 256), 256, 0, stream>>>(mW1, w.wt, 4 * HDc, HDc);
    int T = (Gg / 16) * (HDc / 16);
    k_gemm_bf16_wmma<<<cdiv(T, 8), 256, 0, stream>>>(w.xb, w.wt, z, Gg, 4 * HDc, HDc, T);
    k_bn_relu<<<cdiv(HDc, 256), 256, 0, stream>>>(z, mb1, gamma, beta);
    k_mlp2<<<cdiv(Gg * Cc, 256), 256, 0, stream>>>(z, mW2, mb2, out, Gg * Cc);
}
